// ImplicitGraph_24919400251503
// MI455X (gfx1250) — matrix operand order repaired and checked by reading, not compiled, not run on hardware
//
#include <hip/hip_runtime.h>
#include <hip/hip_bf16.h>
#include <stdint.h>

typedef __attribute__((ext_vector_type(16), aligned(32))) __bf16 v16bf;
typedef __attribute__((ext_vector_type(8)))  float  v8f;

// ---------------------------------------------------------------------------
// Power iteration pieces
// ---------------------------------------------------------------------------
__global__ __launch_bounds__(256) void k_init_v(float* v, int n, float val) {
  int i = blockIdx.x * 256 + threadIdx.x;
  if (i < n) v[i] = val;
}

__global__ __launch_bounds__(256) void k_spmv(const float* __restrict__ vals,
                                              const long long* __restrict__ row,
                                              const long long* __restrict__ col,
                                              const float* __restrict__ v,
                                              float* __restrict__ w, int E) {
  int e = blockIdx.x * 256 + threadIdx.x;
  if (e < E) {
    unsafeAtomicAdd(&w[(int)row[e]], vals[e] * v[(int)col[e]]);
  }
}

__global__ __launch_bounds__(256) void k_norm2(const float* __restrict__ x,
                                               float* __restrict__ ns, int n) {
  __shared__ float sm[256];
  int i = blockIdx.x * 256 + threadIdx.x;
  float val = (i < n) ? x[i] : 0.0f;
  sm[threadIdx.x] = val * val;
  __syncthreads();
  for (int s = 128; s > 0; s >>= 1) {
    if (threadIdx.x < s) sm[threadIdx.x] += sm[threadIdx.x + s];
    __syncthreads();
  }
  if (threadIdx.x == 0) unsafeAtomicAdd(ns, sm[0]);
}

// reads w, zeroes it for the next spmv round (saves one memset node/iter)
__global__ __launch_bounds__(256) void k_update_v(float* __restrict__ w,
                                                  const float* __restrict__ ns,
                                                  float* __restrict__ v, int n) {
  int i = blockIdx.x * 256 + threadIdx.x;
  if (i < n) {
    float x = w[i];
    w[i] = 0.0f;
    v[i] = x / (sqrtf(*ns) + 1e-12f);
  }
}

__global__ void k_rho(const float* __restrict__ ns, float* __restrict__ rho) {
  *rho = sqrtf(*ns) + 1e-5f;  // ||A v|| + EPS
}

// ---------------------------------------------------------------------------
// Row-wise L1-ball projection of W (128x128) -> f32 W_proj.
// One block of 128 threads per row; bitonic sort + Hillis-Steele scan in LDS.
// ---------------------------------------------------------------------------
__global__ __launch_bounds__(128) void k_proj(const float* __restrict__ W,
                                              const float* __restrict__ rho,
                                              float* __restrict__ Wp) {
  const int row = blockIdx.x;
  const int tid = threadIdx.x;
  const float vrad = 0.99f / (*rho);  // KAPPA / A_rho

  __shared__ float s[128];
  __shared__ float cs[128];
  __shared__ int cnt;

  float wv = W[row * 128 + tid];
  float av = fabsf(wv);
  s[tid] = av;
  if (tid == 0) cnt = 0;
  __syncthreads();

  // bitonic sort, descending
  for (int k = 2; k <= 128; k <<= 1) {
    for (int j = k >> 1; j > 0; j >>= 1) {
      int ixj = tid ^ j;
      if (ixj > tid) {
        float x = s[tid], y = s[ixj];
        bool desc = ((tid & k) == 0);
        if (desc ? (x < y) : (x > y)) { s[tid] = y; s[ixj] = x; }
      }
      __syncthreads();
    }
  }

  // inclusive prefix sum of sorted |W|
  cs[tid] = s[tid];
  __syncthreads();
  for (int off = 1; off < 128; off <<= 1) {
    float t = (tid >= off) ? cs[tid - off] : 0.0f;
    __syncthreads();
    cs[tid] += t;
    __syncthreads();
  }
  float row_l1 = cs[127];

  // support size rho = sum(s*k > cs - v)
  int flag = (s[tid] * (float)(tid + 1) > (cs[tid] - vrad)) ? 1 : 0;
  atomicAdd(&cnt, flag);
  __syncthreads();
  int rc = cnt;
  float theta = (rc > 0) ? (cs[rc - 1] - vrad) / (float)rc : 0.0f;

  float pw = copysignf(fmaxf(av - theta, 0.0f), wv);
  Wp[row * 128 + tid] = (row_l1 > vrad) ? pw : wv;
}

// ---------------------------------------------------------------------------
// Sparse scatter-add: Y[row[e], :] += vals[e] * X[col[e], :]
// 2 edges x 128 columns per 256-thread block; f32 atomics resolve in L2.
// ---------------------------------------------------------------------------
__global__ __launch_bounds__(256) void k_spmm(const float* __restrict__ vals,
                                              const long long* __restrict__ row,
                                              const long long* __restrict__ col,
                                              const float* __restrict__ X,
                                              float* __restrict__ Y, int E) {
  int t = blockIdx.x * 256 + threadIdx.x;
  int e = t >> 7;
  int c = t & 127;
  if (e < E) {
    int r = (int)row[e], cl = (int)col[e];
    unsafeAtomicAdd(&Y[r * 128 + c], vals[e] * X[cl * 128 + c]);
  }
}

// ---------------------------------------------------------------------------
// Pack a row-major f32 128x128 matrix into bf16 WMMA-B fragment order:
// dst[(((ks*8 + j)*32 + lane)*16 + t)] = src[(ks*32 + (lane>>4)*16 + t)*128
//                                            + j*16 + (lane&15)]
// so each lane's 16 fragment values are one contiguous, 32B-aligned run.
// ---------------------------------------------------------------------------
__global__ __launch_bounds__(256) void k_pack_bf16(const float* __restrict__ src,
                                                   __bf16* __restrict__ dst) {
  int i = blockIdx.x * 256 + threadIdx.x;   // 0..16383
  int tt   = i & 15;
  int lane = (i >> 4) & 31;
  int j    = (i >> 9) & 7;
  int ks   = i >> 12;
  int m    = lane & 15;
  int hf   = lane >> 4;
  int K    = ks * 32 + hf * 16 + tt;
  int colc = j * 16 + m;
  dst[i] = (__bf16)src[K * 128 + colc];
}

// ---------------------------------------------------------------------------
// Dense GEMM: C[nrows x 128] = A[nrows x 128] (f32, cvt->bf16 inline)
//                               @ B (bf16, pre-swizzled fragment layout).
// One wave per 16-row strip; 8 f32 accumulators cover all 128 columns, so the
// A fragment is loaded once per K-step and reused by 8 WMMAs.
// Per K-step: 4x b128 A loads + 8x cvt_pk + 8x (1x 32B B load + 1x WMMA).
// ---------------------------------------------------------------------------
__global__ __launch_bounds__(32) void k_gemm_bf16(const float* __restrict__ A,
                                                  const __bf16* __restrict__ B16,
                                                  float* __restrict__ C, int nrows) {
  const int lane = threadIdx.x & 31;
  const int m = lane & 15;
  const int hf = lane >> 4;
  const long long r0 = (long long)blockIdx.x * 16;
  const v16bf* __restrict__ bfrag = (const v16bf*)B16;

  const v8f zero = {0.f, 0.f, 0.f, 0.f, 0.f, 0.f, 0.f, 0.f};
  v8f acc[8];
#pragma unroll
  for (int j = 0; j < 8; ++j) acc[j] = zero;

#pragma unroll
  for (int ks = 0; ks < 4; ++ks) {
    const int k0 = ks * 32;
    // A fragment (16x32 bf16): lane half hf holds K = k0+hf*16..+15 in VGPR0-7
    // (contiguous order, matching the packed B fragments).
    const float4* ap = (const float4*)(A + (r0 + m) * 128 + k0 + hf * 16);
    float4 p0 = ap[0], p1 = ap[1];   // K = k0 + hf*16 + 0..7
    float4 q0 = ap[2], q1 = ap[3];   // K = k0 + hf*16 + 8..15
    v16bf a;
    a[0]  = (__bf16)p0.x; a[1]  = (__bf16)p0.y; a[2]  = (__bf16)p0.z; a[3]  = (__bf16)p0.w;
    a[4]  = (__bf16)p1.x; a[5]  = (__bf16)p1.y; a[6]  = (__bf16)p1.z; a[7]  = (__bf16)p1.w;
    a[8]  = (__bf16)q0.x; a[9]  = (__bf16)q0.y; a[10] = (__bf16)q0.z; a[11] = (__bf16)q0.w;
    a[12] = (__bf16)q1.x; a[13] = (__bf16)q1.y; a[14] = (__bf16)q1.z; a[15] = (__bf16)q1.w;

#pragma unroll
    for (int j = 0; j < 8; ++j) {
      v16bf b = bfrag[(ks * 8 + j) * 32 + lane];   // one aligned 32B load
      acc[j] = __builtin_amdgcn_wmma_f32_16x16x32_bf16(
          /*neg_a=*/false, a, /*neg_b=*/false, b,
          /*c_mod=*/(short)0, acc[j], /*reuse_a=*/false, /*reuse_b=*/false);
    }
  }

  // C/D layout: lanes 0-15 -> rows r0+0..7 (VGPR0..7), lanes 16-31 -> r0+8..15
#pragma unroll
  for (int j = 0; j < 8; ++j) {
    float* cp = C + (r0 + hf * 8) * 128 + j * 16 + m;
#pragma unroll
    for (int r = 0; r < 8; ++r) cp[r * 128] = acc[j][r];
  }
}

__global__ __launch_bounds__(256) void k_relu(float* __restrict__ x, int n) {
  int i = blockIdx.x * 256 + threadIdx.x;
  if (i < n) x[i] = fmaxf(x[i], 0.0f);
}

// ---------------------------------------------------------------------------
// Host-side orchestration (all on `stream`, graph-capture safe)
// ---------------------------------------------------------------------------
extern "C" void kernel_launch(void* const* d_in, const int* in_sizes, int n_in,
                              void* d_out, int out_size, void* d_ws, size_t ws_size,
                              hipStream_t stream) {
  (void)n_in; (void)out_size; (void)ws_size;
  const float*     X_pre = (const float*)d_in[0];       // [N,128]
  const float*     U     = (const float*)d_in[1];       // [N,128]
  const float*     W     = (const float*)d_in[2];       // [128,128]
  const float*     Bm    = (const float*)d_in[3];       // [128,128]
  const float*     evals = (const float*)d_in[4];       // [E]
  const long long* eidx  = (const long long*)d_in[5];   // [2,E] int64

  const int N = in_sizes[1] / 128;
  const int E = in_sizes[4];
  const long long* erow = eidx;
  const long long* ecol = eidx + E;
  float* out = (float*)d_out;

  // workspace layout (256B aligned)
  char* ws = (char*)d_ws;
  size_t off = 0;
  auto take = [&](size_t bytes) -> char* {
    char* p = ws + off;
    off += (bytes + 255) & ~(size_t)255;
    return p;
  };
  float*  v    = (float*)take((size_t)N * 4);
  float*  wv   = (float*)take((size_t)N * 4);
  float*  ns   = (float*)take(4);
  float*  rho  = (float*)take(4);
  float*  tmp  = (float*)take((size_t)N * 128 * 4);   // AU, then reused for XW
  float*  Wp   = (float*)take(128 * 128 * 4);
  __bf16* B16  = (__bf16*)take(128 * 128 * 2);        // swizzled fragments
  __bf16* Wp16 = (__bf16*)take(128 * 128 * 2);        // swizzled fragments

  const int nb_n   = (N + 255) / 256;
  const int nb_e   = (E + 255) / 256;
  const int nb_sp  = (E + 1) / 2;                  // 2 edges * 128 cols / block
  const int nb_out = (N * 128 + 255) / 256;

  // ---- A_U_B: tmp = A @ U ; out = tmp @ B (bf16 WMMA, f32 accumulate) ----
  hipMemsetAsync(tmp, 0, (size_t)N * 128 * 4, stream);
  k_spmm<<<nb_sp, 256, 0, stream>>>(evals, erow, ecol, U, tmp, E);
  k_pack_bf16<<<64, 256, 0, stream>>>(Bm, B16);
  k_gemm_bf16<<<N / 16, 32, 0, stream>>>(tmp, B16, out, N);

  // ---- spectral radius via power iteration ----
  k_init_v<<<nb_n, 256, 0, stream>>>(v, N, 1.0f / sqrtf((float)N));
  hipMemsetAsync(wv, 0, (size_t)N * 4, stream);
  for (int it = 0; it < 100; ++it) {
    hipMemsetAsync(ns, 0, 4, stream);
    k_spmv<<<nb_e, 256, 0, stream>>>(evals, erow, ecol, v, wv, E);
    k_norm2<<<nb_n, 256, 0, stream>>>(wv, ns, N);
    k_update_v<<<nb_n, 256, 0, stream>>>(wv, ns, v, N);  // also re-zeroes wv
  }
  hipMemsetAsync(ns, 0, 4, stream);
  k_spmv<<<nb_e, 256, 0, stream>>>(evals, erow, ecol, v, wv, E);
  k_norm2<<<nb_n, 256, 0, stream>>>(wv, ns, N);
  k_rho<<<1, 1, 0, stream>>>(ns, rho);

  // ---- W projection, second GEMM, scatter-add, relu ----
  k_proj<<<128, 128, 0, stream>>>(W, rho, Wp);
  k_pack_bf16<<<64, 256, 0, stream>>>(Wp, Wp16);
  k_gemm_bf16<<<N / 16, 32, 0, stream>>>(X_pre, Wp16, tmp, N);   // tmp = X_pre @ W_proj
  k_spmm<<<nb_sp, 256, 0, stream>>>(evals, erow, ecol, tmp, out, E); // out += A @ tmp
  k_relu<<<nb_out, 256, 0, stream>>>(out, N * 128);
}
